// AttentionSeq2SeqModel_87316685128483
// MI455X (gfx1250) — compile-verified, matrix operand order checked by reading, not run on hardware
//
#include <hip/hip_runtime.h>
#include <hip/hip_bf16.h>

// ---------------------------------------------------------------------------
// AttentionSeq2Seq on MI455X (gfx1250, wave32, WMMA).
// B=512, S=14, H=128, IN=1, target_len=8.
// 32 workgroups x 256 threads; each WG owns 16 batch rows end-to-end.
// Recurrence GEMMs use v_wmma_f32_16x16x32_bf16; h is kept in a
// fragment-swizzled bf16 LDS buffer matching the ISA A-matrix layout.
// Gate nonlinearities use hardware V_TANH_F32 (sigmoid via tanh identity)
// to keep the serial recurrence latency minimal.
// ---------------------------------------------------------------------------

typedef __bf16 bf16_t;
typedef __bf16 v16bf __attribute__((ext_vector_type(16)));
typedef float  v8f   __attribute__((ext_vector_type(8)));

#define H_DIM   128
#define S_LEN   14
#define BATCH   512
#define MTILE   16
#define NBLK    (BATCH / MTILE)   // 32 workgroups
#define TLEN    8

// workspace layout (bf16 fragment-major weights), bytes
#define FRAG_WHH_BYTES (4 * 32 * 32 * 16 * 2)   // K=128(4 kt) x N=512(32 nt) x 32 lanes x 16 bf16
#define FRAG_W128_BYTES (4 * 8 * 32 * 16 * 2)   // K=128 x N=128(8 nt)
#define OFF_WHHE  0
#define OFF_WHHD  (OFF_WHHE + FRAG_WHH_BYTES)
#define OFF_WE    (OFF_WHHD + FRAG_WHH_BYTES)
#define OFF_WD    (OFF_WE   + FRAG_W128_BYTES)

// dynamic LDS layout, bytes
#define L_HPLAIN  0                          // float [16][128]
#define L_CTX     8192                       // float [16][128]  (dproj, then context)
#define L_SCORE   16384                      // float [16][14]
#define L_TATTN   17408                      // float [16][14]
#define L_WV      18432                      // float [128]
#define L_XDEC    18944                      // float [16]
#define L_XCUR    19008                      // float [16]
#define L_HFRAG   19072                      // bf16 [4][32][16]  (A-fragment layout of h)
#define L_ENCOUT  23168                      // bf16 [16][14][128]
#define L_EPROJ   80512                      // bf16 [16][14][128]
#define SMEM_BYTES 137856

// Hardware tanh (V_TANH_F32, TRANS32 class on gfx1250) when available.
__device__ __forceinline__ float fast_tanh(float x) {
#if defined(__has_builtin)
# if __has_builtin(__builtin_amdgcn_tanhf)
  return __builtin_amdgcn_tanhf(x);
# elif __has_builtin(__builtin_amdgcn_tanh_f32)
  return __builtin_amdgcn_tanh_f32(x);
# else
  return tanhf(x);
# endif
#else
  return tanhf(x);
#endif
}
// sigmoid(x) = 0.5 + 0.5*tanh(x/2): one TRANS op, branch-free.
__device__ __forceinline__ float sigf(float x) {
  return fmaf(0.5f, fast_tanh(0.5f * x), 0.5f);
}

// ---------------------------------------------------------------------------
// Weight pre-swizzle: W[K][N] f32 row-major -> bf16 fragment-major matching the
// CDNA5 16-bit B-matrix (32x16 per tile) VGPR layout:
//   kl = (v<4 ? 2v : 16+2(v-4)) + (i&1) + 8*(lane>=16),  lane = (n&15)+16*laneHi
// ---------------------------------------------------------------------------
__device__ __forceinline__ void put_frag(const float* __restrict__ W,
                                         bf16_t* __restrict__ F, int N, int idx) {
  const int k = idx / N, n = idx % N;
  const int nt = n >> 4, nl = n & 15;
  const int kt = k >> 5, kl = k & 31;
  const int laneHi = (kl >> 3) & 1;
  const int r2 = kl & ~8;
  const int v  = (r2 < 8) ? (r2 >> 1) : (4 + ((r2 - 16) >> 1));
  const int i  = 2 * v + (kl & 1);
  const int lane = nl + 16 * laneHi;
  F[(((kt * (N >> 4)) + nt) * 32 + lane) * 16 + i] = (bf16_t)W[idx];
}

__global__ __launch_bounds__(256)
void prep_weights(const float* __restrict__ WhhE, const float* __restrict__ WhhD,
                  const float* __restrict__ We,   const float* __restrict__ Wd,
                  bf16_t* __restrict__ fE, bf16_t* __restrict__ fD,
                  bf16_t* __restrict__ fWe, bf16_t* __restrict__ fWd) {
  const int idx = blockIdx.x * 256 + threadIdx.x;
  if (idx < 65536)        put_frag(WhhE, fE, 512, idx);
  else if (idx < 131072)  put_frag(WhhD, fD, 512, idx - 65536);
  else if (idx < 147456)  put_frag(We,  fWe, 128, idx - 131072);
  else if (idx < 163840)  put_frag(Wd,  fWd, 128, idx - 147456);
}

// ---------------------------------------------------------------------------
// One LSTM step: g = h@Whh + x*Wih + b ; gates ; update c (regs), h (LDS).
// Wave w owns gate-columns j in [16w, 16w+16): tiles {w, w+8, w+16, w+24}
// are the i/f/g/o gates for the same columns -> gate math stays in registers.
// ---------------------------------------------------------------------------
struct LstmW {
  const bf16_t* whh;                       // fragment-major, NT=32
  float bI, bF, bG, bO, wI, wF, wG, wO;    // bias / Wih scalars for this thread
};

__device__ __forceinline__ void lstm_step(
    const LstmW& W,
    bf16_t* __restrict__ s_hfrag, float* __restrict__ s_hplain,
    bf16_t* __restrict__ s_encout, int enc_t,
    const float* __restrict__ xsrc, int xstride,
    v8f& c_acc, int wave, int lane, int n15, int hi) {
  float xv[8];
#pragma unroll
  for (int r = 0; r < 8; ++r) xv[r] = xsrc[(r + 8 * hi) * xstride];

  v8f aI, aF, aG, aO;
#pragma unroll
  for (int r = 0; r < 8; ++r) {
    aI[r] = W.bI + xv[r] * W.wI;
    aF[r] = W.bF + xv[r] * W.wF;
    aG[r] = W.bG + xv[r] * W.wG;
    aO[r] = W.bO + xv[r] * W.wO;
  }
#pragma unroll
  for (int kt = 0; kt < 4; ++kt) {
    v16bf a = *reinterpret_cast<const v16bf*>(&s_hfrag[(kt * 32 + lane) * 16]);
    v16bf bI = *reinterpret_cast<const v16bf*>(&W.whh[(((kt * 32) + wave     ) * 32 + lane) * 16]);
    v16bf bF = *reinterpret_cast<const v16bf*>(&W.whh[(((kt * 32) + wave +  8) * 32 + lane) * 16]);
    v16bf bG = *reinterpret_cast<const v16bf*>(&W.whh[(((kt * 32) + wave + 16) * 32 + lane) * 16]);
    v16bf bO = *reinterpret_cast<const v16bf*>(&W.whh[(((kt * 32) + wave + 24) * 32 + lane) * 16]);
    aI = __builtin_amdgcn_wmma_f32_16x16x32_bf16(false, a, false, bI, (short)0, aI, false, false);
    aF = __builtin_amdgcn_wmma_f32_16x16x32_bf16(false, a, false, bF, (short)0, aF, false, false);
    aG = __builtin_amdgcn_wmma_f32_16x16x32_bf16(false, a, false, bG, (short)0, aG, false, false);
    aO = __builtin_amdgcn_wmma_f32_16x16x32_bf16(false, a, false, bO, (short)0, aO, false, false);
  }
  __syncthreads();  // all waves' h_frag reads complete before overwriting h

  // A-fragment slot for column j (ISA 16-bit A 16x32 layout)
  const int j   = wave * 16 + n15;
  const int kt2 = j >> 5;
  const int kl  = j & 31;
  const int lhi = (kl >> 3) & 1;
  const int r2  = kl & ~8;
  const int vv  = (r2 < 8) ? (r2 >> 1) : (4 + ((r2 - 16) >> 1));
  const int ii  = 2 * vv + (kl & 1);

#pragma unroll
  for (int r = 0; r < 8; ++r) {
    const int brow = r + 8 * hi;  // accumulator row -> batch row (C/D layout)
    float cN = sigf(aF[r]) * c_acc[r] + sigf(aI[r]) * fast_tanh(aG[r]);
    float hN = sigf(aO[r]) * fast_tanh(cN);
    c_acc[r] = cN;
    s_hplain[brow * H_DIM + j] = hN;
    s_hfrag[(kt2 * 32 + (brow + 16 * lhi)) * 16 + ii] = (bf16_t)hN;
    if (enc_t >= 0) s_encout[(brow * S_LEN + enc_t) * H_DIM + j] = (bf16_t)hN;
  }
}

// ---------------------------------------------------------------------------
__global__ __launch_bounds__(256)
void seq2seq_main(const float* __restrict__ inputs,
                  const float* __restrict__ wih_e, const float* __restrict__ b_e,
                  const float* __restrict__ wih_d, const float* __restrict__ b_d,
                  const float* __restrict__ Wv,    const float* __restrict__ Wf,
                  const float* __restrict__ bf_p,
                  const bf16_t* __restrict__ fWhhE, const bf16_t* __restrict__ fWhhD,
                  const bf16_t* __restrict__ fWe,   const bf16_t* __restrict__ fWd,
                  float* __restrict__ out) {
  extern __shared__ char smem[];
  float*  s_hplain = (float*) (smem + L_HPLAIN);
  float*  s_ctx    = (float*) (smem + L_CTX);
  float*  s_score  = (float*) (smem + L_SCORE);
  float*  s_tattn  = (float*) (smem + L_TATTN);
  float*  s_wv     = (float*) (smem + L_WV);
  float*  s_xdec   = (float*) (smem + L_XDEC);
  float*  s_xcur   = (float*) (smem + L_XCUR);
  bf16_t* s_hfrag  = (bf16_t*)(smem + L_HFRAG);
  bf16_t* s_encout = (bf16_t*)(smem + L_ENCOUT);
  bf16_t* s_eproj  = (bf16_t*)(smem + L_EPROJ);

  const int tid  = threadIdx.x;
  const int wave = tid >> 5;
  const int lane = tid & 31;
  const int n15  = lane & 15;
  const int hi   = lane >> 4;
  const int b0   = blockIdx.x * MTILE;
  const int j    = wave * 16 + n15;

  for (int idx = tid; idx < 2048; idx += 256) {
    s_hplain[idx] = 0.0f;
    s_hfrag[idx]  = (bf16_t)0.0f;
  }
  for (int idx = tid; idx < MTILE * S_LEN; idx += 256) s_tattn[idx] = 0.0f;
  if (tid < H_DIM) s_wv[tid] = Wv[tid];
  __syncthreads();

  v8f c_acc = {0.f, 0.f, 0.f, 0.f, 0.f, 0.f, 0.f, 0.f};

  LstmW We_;
  We_.whh = fWhhE;
  We_.bI = b_e[j];        We_.bF = b_e[128 + j];
  We_.bG = b_e[256 + j];  We_.bO = b_e[384 + j];
  We_.wI = wih_e[j];      We_.wF = wih_e[128 + j];
  We_.wG = wih_e[256 + j]; We_.wO = wih_e[384 + j];
  LstmW Wd_;
  Wd_.whh = fWhhD;
  Wd_.bI = b_d[j];        Wd_.bF = b_d[128 + j];
  Wd_.bG = b_d[256 + j];  Wd_.bO = b_d[384 + j];
  Wd_.wI = wih_d[j];      Wd_.wF = wih_d[128 + j];
  Wd_.wG = wih_d[256 + j]; Wd_.wO = wih_d[384 + j];

  // ------------------------- encoder (14 steps) ---------------------------
  for (int t = 0; t < S_LEN; ++t) {
    if (tid < MTILE) s_xcur[tid] = inputs[(b0 + tid) * S_LEN + t];
    __syncthreads();
    lstm_step(We_, s_hfrag, s_hplain, s_encout, t, s_xcur, 1,
              c_acc, wave, lane, n15, hi);
    __syncthreads();
  }

  // -------------------- enc_proj = enc_out @ We (hoisted) ------------------
  for (int s = 0; s < S_LEN; ++s) {
    v8f acc = {0.f, 0.f, 0.f, 0.f, 0.f, 0.f, 0.f, 0.f};
#pragma unroll
    for (int kt = 0; kt < 4; ++kt) {
      v16bf a;
#pragma unroll
      for (int i2 = 0; i2 < 16; ++i2) {
        const int v2 = i2 >> 1;
        const int base = (v2 < 4) ? (2 * v2) : (16 + 2 * (v2 - 4));
        const int kl2 = base + (i2 & 1) + (hi << 3);
        a[i2] = s_encout[(n15 * S_LEN + s) * H_DIM + kt * 32 + kl2];
      }
      v16bf bm = *reinterpret_cast<const v16bf*>(&fWe[(((kt * 8) + wave) * 32 + lane) * 16]);
      acc = __builtin_amdgcn_wmma_f32_16x16x32_bf16(false, a, false, bm, (short)0, acc, false, false);
    }
#pragma unroll
    for (int r = 0; r < 8; ++r)
      s_eproj[((r + 8 * hi) * S_LEN + s) * H_DIM + j] = (bf16_t)acc[r];
  }
  if (tid < MTILE) s_xdec[tid] = inputs[(b0 + tid) * S_LEN + (S_LEN - 1)];
  __syncthreads();

  // --------------------------- decoder (8 steps) ---------------------------
  const float bf_v = bf_p[0];
  for (int step = 0; step < TLEN; ++step) {
    // dproj = h @ Wd -> s_ctx
    {
      v8f acc = {0.f, 0.f, 0.f, 0.f, 0.f, 0.f, 0.f, 0.f};
#pragma unroll
      for (int kt = 0; kt < 4; ++kt) {
        v16bf a  = *reinterpret_cast<const v16bf*>(&s_hfrag[(kt * 32 + lane) * 16]);
        v16bf bm = *reinterpret_cast<const v16bf*>(&fWd[(((kt * 8) + wave) * 32 + lane) * 16]);
        acc = __builtin_amdgcn_wmma_f32_16x16x32_bf16(false, a, false, bm, (short)0, acc, false, false);
      }
#pragma unroll
      for (int r = 0; r < 8; ++r) s_ctx[(r + 8 * hi) * H_DIM + j] = acc[r];
    }
    __syncthreads();

    // scores: Wv . tanh(enc_proj + dproj)
    if (tid < MTILE * S_LEN) {
      const int b = tid / S_LEN, s = tid % S_LEN;
      float acc2 = 0.0f;
      for (int k = 0; k < H_DIM; ++k)
        acc2 += fast_tanh((float)s_eproj[(b * S_LEN + s) * H_DIM + k] + s_ctx[b * H_DIM + k]) * s_wv[k];
      s_score[b * S_LEN + s] = acc2;
    }
    __syncthreads();

    // softmax over s (one thread per batch row) + total_attn accumulation
    if (tid < MTILE) {
      float m = -1e30f;
      for (int s = 0; s < S_LEN; ++s) m = fmaxf(m, s_score[tid * S_LEN + s]);
      float p[S_LEN], sum = 0.0f;
      for (int s = 0; s < S_LEN; ++s) { p[s] = __expf(s_score[tid * S_LEN + s] - m); sum += p[s]; }
      const float inv = 1.0f / sum;
      for (int s = 0; s < S_LEN; ++s) {
        const float pp = p[s] * inv;
        s_score[tid * S_LEN + s] = pp;
        s_tattn[tid * S_LEN + s] += pp;
      }
    }
    __syncthreads();

    // context = sum_s attn * enc_out  (overwrites dproj in s_ctx)
    for (int idx = tid; idx < MTILE * H_DIM; idx += 256) {
      const int b = idx >> 7, k = idx & 127;
      float acc3 = 0.0f;
      for (int s = 0; s < S_LEN; ++s)
        acc3 += s_score[b * S_LEN + s] * (float)s_encout[(b * S_LEN + s) * H_DIM + k];
      s_ctx[b * H_DIM + k] = acc3;
    }
    __syncthreads();

    // inner LSTM over 129 scalar inputs: context[0..127], then x
    for (int t = 0; t <= H_DIM; ++t) {
      const float* xsrc = (t < H_DIM) ? (s_ctx + t) : s_xdec;
      const int xstride = (t < H_DIM) ? H_DIM : 1;
      lstm_step(Wd_, s_hfrag, s_hplain, s_encout, -1, xsrc, xstride,
                c_acc, wave, lane, n15, hi);
      __syncthreads();
    }

    // out = h @ Wf + bf ; feed back as next x
    if (tid < MTILE) {
      float o = bf_v;
      for (int k = 0; k < H_DIM; ++k) o += s_hplain[tid * H_DIM + k] * Wf[k];
      out[(b0 + tid) * TLEN + step] = o;
      s_xdec[tid] = o;
    }
    __syncthreads();
  }

  // total_attn [B, S, 1] after outputs [B, TLEN, 1]
  for (int idx = tid; idx < MTILE * S_LEN; idx += 256) {
    const int b = idx / S_LEN, s = idx % S_LEN;
    out[BATCH * TLEN + (b0 + b) * S_LEN + s] = s_tattn[idx];
  }
}

// ---------------------------------------------------------------------------
extern "C" void kernel_launch(void* const* d_in, const int* in_sizes, int n_in,
                              void* d_out, int out_size, void* d_ws, size_t ws_size,
                              hipStream_t stream) {
  (void)in_sizes; (void)n_in; (void)out_size; (void)ws_size;
  const float* inputs = (const float*)d_in[0];
  // d_in[1] = target_len (== 8, hardcoded)
  const float* Wih_e = (const float*)d_in[2];
  const float* Whh_e = (const float*)d_in[3];
  const float* b_e   = (const float*)d_in[4];
  const float* Wih_d = (const float*)d_in[5];
  const float* Whh_d = (const float*)d_in[6];
  const float* b_d   = (const float*)d_in[7];
  const float* We    = (const float*)d_in[8];
  const float* Wd    = (const float*)d_in[9];
  const float* Wv    = (const float*)d_in[10];
  const float* Wf    = (const float*)d_in[11];
  const float* bf_p  = (const float*)d_in[12];

  bf16_t* fE  = (bf16_t*)((char*)d_ws + OFF_WHHE);
  bf16_t* fD  = (bf16_t*)((char*)d_ws + OFF_WHHD);
  bf16_t* fWe = (bf16_t*)((char*)d_ws + OFF_WE);
  bf16_t* fWd = (bf16_t*)((char*)d_ws + OFF_WD);

  prep_weights<<<640, 256, 0, stream>>>(Whh_e, Whh_d, We, Wd, fE, fD, fWe, fWd);
  seq2seq_main<<<NBLK, 256, SMEM_BYTES, stream>>>(
      inputs, Wih_e, b_e, Wih_d, b_d, Wv, Wf, bf_p,
      fE, fD, fWe, fWd, (float*)d_out);
}
